// AdaptiveSparseAttention_8555574854168
// MI455X (gfx1250) — matrix operand order repaired and checked by reading, not compiled, not run on hardware
//
#include <hip/hip_runtime.h>
#include <math.h>

// Problem constants (fixed by the reference)
#define HIDDEN   1024
#define HEADS    16
#define HEAD_DIM 64
#define SEQ      2048
#define BATCH    2
#define TOKENS   (BATCH * SEQ)   // 4096
#define KKEEP    1024            // int(S * (1 - 0.5))

typedef __attribute__((ext_vector_type(16))) __bf16 v16bf;
typedef __attribute__((ext_vector_type(8)))  float  v8f;
typedef __attribute__((ext_vector_type(4)))  unsigned int u32x4;
typedef int v4i_ __attribute__((vector_size(4 * sizeof(int))));

union Frag16 { unsigned int u[8]; u32x4 q[2]; v16bf v; };

// ---- CDNA5 async global->LDS path (guarded; fallback = reg-staged copy) ----
#if __has_builtin(__builtin_amdgcn_global_load_async_to_lds_b128) && \
    __has_builtin(__builtin_amdgcn_s_wait_asynccnt)
#define ASYNC_LDS 1
#else
#define ASYNC_LDS 0
#endif

#if ASYNC_LDS
__device__ __forceinline__ void async_cp16(const void* g, void* l) {
  __builtin_amdgcn_global_load_async_to_lds_b128(
      (__attribute__((address_space(1))) v4i_*)g,
      (__attribute__((address_space(3))) v4i_*)l, 0, 0);
}
#endif

__device__ __forceinline__ unsigned short f32_to_bf16(float f) {
  unsigned int u = __float_as_uint(f);
  u += 0x7FFFu + ((u >> 16) & 1u);   // round-to-nearest-even
  return (unsigned short)(u >> 16);
}
__device__ __forceinline__ unsigned int pack2_bf16(float lo, float hi) {
#if __has_builtin(__builtin_amdgcn_cvt_pk_bf16_f32)
  typedef __attribute__((ext_vector_type(2))) __bf16 v2bf;
  v2bf r = __builtin_amdgcn_cvt_pk_bf16_f32(lo, hi);
  return __builtin_bit_cast(unsigned int, r);
#else
  return (unsigned int)f32_to_bf16(lo) | ((unsigned int)f32_to_bf16(hi) << 16);
#endif
}
// Monotone float<->uint key (total order matching float compare for non-NaN)
__device__ __forceinline__ unsigned int mono_key(float f) {
  unsigned int u = __float_as_uint(f);
  return (u & 0x80000000u) ? ~u : (u | 0x80000000u);
}
__device__ __forceinline__ float key_to_f(unsigned int k) {
  unsigned int u = (k & 0x80000000u) ? (k ^ 0x80000000u) : ~k;
  return __uint_as_float(u);
}

// ---------------------------------------------------------------------------
// fp32 -> bf16 elementwise conversion
// ---------------------------------------------------------------------------
__global__ void cvt_f32_to_bf16_k(const float* __restrict__ in,
                                  unsigned short* __restrict__ out, int n) {
  int i = blockIdx.x * blockDim.x + threadIdx.x;
  if (i < n) out[i] = f32_to_bf16(in[i]);
}

// ---------------------------------------------------------------------------
// Tiled WMMA GEMM:  C[M,N] = A[M,K] * W[N,K]^T   (A, W bf16, accum f32)
// 256 threads (8 waves), 128x128 tile, K staged 32 wide, double-buffered LDS.
// MODE 0: fp32 row-major out + bias   MODE 1: bf16 [B,H,S,D]   MODE 2: [B,H,D,S]
// ---------------------------------------------------------------------------
template <int MODE>
__global__ __launch_bounds__(256)
void wmma_gemm_nt_k(const unsigned short* __restrict__ A,
                    const unsigned short* __restrict__ W,
                    const float* __restrict__ bias,
                    void* __restrict__ out,
                    int M, int N, int K) {
  __shared__ unsigned short As[2][128 * 32];
  __shared__ unsigned short Bs[2][128 * 32];

  const int tid  = threadIdx.x;
  const int w    = tid >> 5;
  const int lane = tid & 31;
  const int half = lane >> 4;
  const int l16  = lane & 15;
  const int m0   = blockIdx.y * 128;
  const int n0   = blockIdx.x * 128;
  const int mp   = (w & 3) * 2;   // wave owns local m-tiles {mp, mp+1}
  const int ng   = (w >> 2) * 4;  // wave owns local n-tiles {ng .. ng+3}

  // Cooperative panel-copy addressing: 512 u32x4 per 128x32 bf16 panel
  const int idx0 = tid, idx1 = tid + 256;
  const unsigned short* gA0 = A + (size_t)(m0 + (idx0 >> 2)) * K + (idx0 & 3) * 8;
  const unsigned short* gA1 = A + (size_t)(m0 + (idx1 >> 2)) * K + (idx1 & 3) * 8;
  const unsigned short* gB0 = W + (size_t)(n0 + (idx0 >> 2)) * K + (idx0 & 3) * 8;
  const unsigned short* gB1 = W + (size_t)(n0 + (idx1 >> 2)) * K + (idx1 & 3) * 8;

  v8f c[2][4];
#pragma unroll
  for (int i = 0; i < 2; ++i)
#pragma unroll
    for (int j = 0; j < 4; ++j)
      c[i][j] = (v8f){0.f, 0.f, 0.f, 0.f, 0.f, 0.f, 0.f, 0.f};

  const int nsteps = K >> 5;

#if ASYNC_LDS
  // ---- async double-buffer: panels stream straight into LDS --------------
  {
    async_cp16(gA0, &As[0][idx0 * 8]);
    async_cp16(gA1, &As[0][idx1 * 8]);
    async_cp16(gB0, &Bs[0][idx0 * 8]);
    async_cp16(gB1, &Bs[0][idx1 * 8]);
    __builtin_amdgcn_s_wait_asynccnt(0);
    __syncthreads();
  }
#else
  u32x4 ra0, ra1, rb0, rb1;
  ra0 = *(const u32x4*)gA0; ra1 = *(const u32x4*)gA1;
  rb0 = *(const u32x4*)gB0; rb1 = *(const u32x4*)gB1;
  *(u32x4*)&As[0][idx0 * 8] = ra0; *(u32x4*)&As[0][idx1 * 8] = ra1;
  *(u32x4*)&Bs[0][idx0 * 8] = rb0; *(u32x4*)&Bs[0][idx1 * 8] = rb1;
  __syncthreads();
#endif

  for (int kt = 0; kt < nsteps; ++kt) {
    const int cur = kt & 1;
    const int nxt = cur ^ 1;
    const bool have_next = (kt + 1) < nsteps;
    const int knext = (kt + 1) << 5;

#if ASYNC_LDS
    if (have_next) {   // fill the idle buffer while WMMAs run on 'cur'
      async_cp16(gA0 + knext, &As[nxt][idx0 * 8]);
      async_cp16(gA1 + knext, &As[nxt][idx1 * 8]);
      async_cp16(gB0 + knext, &Bs[nxt][idx0 * 8]);
      async_cp16(gB1 + knext, &Bs[nxt][idx1 * 8]);
    }
#else
    if (have_next) {   // global loads in flight during the WMMAs
      ra0 = *(const u32x4*)(gA0 + knext); ra1 = *(const u32x4*)(gA1 + knext);
      rb0 = *(const u32x4*)(gB0 + knext); rb1 = *(const u32x4*)(gB1 + knext);
    }
#endif

    // Fragments: two contiguous b128 runs per lane (A & B layouts, CDNA5 ISA)
    const char* Ab = (const char*)As[cur];
    const char* Bb = (const char*)Bs[cur];
    Frag16 af[2], bf[4];
#pragma unroll
    for (int i = 0; i < 2; ++i) {
      int m = (mp + i) * 16 + l16;
      af[i].q[0] = *(const u32x4*)(Ab + m * 64 + 16 * half);
      af[i].q[1] = *(const u32x4*)(Ab + m * 64 + 32 + 16 * half);
    }
#pragma unroll
    for (int j = 0; j < 4; ++j) {
      int n = (ng + j) * 16 + l16;
      bf[j].q[0] = *(const u32x4*)(Bb + n * 64 + 16 * half);
      bf[j].q[1] = *(const u32x4*)(Bb + n * 64 + 32 + 16 * half);
    }
#pragma unroll
    for (int i = 0; i < 2; ++i)
#pragma unroll
      for (int j = 0; j < 4; ++j)
        c[i][j] = __builtin_amdgcn_wmma_f32_16x16x32_bf16(
            false, af[i].v, false, bf[j].v, (short)0, c[i][j], false, false);

#if ASYNC_LDS
    __builtin_amdgcn_s_wait_asynccnt(0);
    __syncthreads();
#else
    __syncthreads();
    if (have_next) {
      *(u32x4*)&As[nxt][idx0 * 8] = ra0; *(u32x4*)&As[nxt][idx1 * 8] = ra1;
      *(u32x4*)&Bs[nxt][idx0 * 8] = rb0; *(u32x4*)&Bs[nxt][idx1 * 8] = rb1;
      __syncthreads();
    }
#endif
  }

  // Epilogue
#pragma unroll
  for (int i = 0; i < 2; ++i) {
#pragma unroll
    for (int j = 0; j < 4; ++j) {
#pragma unroll
      for (int r = 0; r < 8; ++r) {
        int m = m0 + (mp + i) * 16 + r + 8 * half;
        int n = n0 + (ng + j) * 16 + l16;
        float val = c[i][j][r];
        if (MODE == 0) {
          ((float*)out)[(size_t)m * N + n] = val + (bias ? bias[n] : 0.f);
        } else {
          int b_ = m >> 11, s = m & (SEQ - 1);
          int h  = n >> 6,  d = n & (HEAD_DIM - 1);
          if (MODE == 1)  // [B,H,S,D]
            ((unsigned short*)out)[(((size_t)(b_ * HEADS + h) * SEQ + s) *
                                    HEAD_DIM) + d] = f32_to_bf16(val);
          else            // [B,H,D,S]
            ((unsigned short*)out)[(((size_t)(b_ * HEADS + h) * HEAD_DIM + d) *
                                    SEQ) + s] = f32_to_bf16(val);
        }
      }
    }
  }
}

// ---------------------------------------------------------------------------
// Adaptive sparse attention core.
// One workgroup = one (b,h) x 16-query tile. 512 threads = 16 waves.
// Dynamic LDS: 16 x 2048 fp32 score strip (128 KB), reused for partials.
// ---------------------------------------------------------------------------
__global__ __launch_bounds__(512)
void attn_sparse_k(const unsigned short* __restrict__ Q,   // [B*H][S][D] bf16
                   const unsigned short* __restrict__ Km,  // [B*H][S][D] bf16
                   const unsigned short* __restrict__ Vt,  // [B*H][D][S] bf16
                   const float* __restrict__ thr_ptr,
                   unsigned short* __restrict__ attn_out)  // [B*S][1024] bf16
{
  extern __shared__ float sc[];  // [16][SEQ]

  const int tid  = threadIdx.x;
  const int w    = tid >> 5;
  const int lane = tid & 31;
  const int half = lane >> 4;
  const int l16  = lane & 15;
  const int bh   = blockIdx.x >> 7;    // 128 q-tiles per head
  const int qt   = blockIdx.x & 127;

  const char* Qh = (const char*)(Q  + (size_t)bh * SEQ * HEAD_DIM);
  const char* Kh = (const char*)(Km + (size_t)bh * SEQ * HEAD_DIM);
  const char* Vh = (const char*)(Vt + (size_t)bh * HEAD_DIM * SEQ);

  // ---- Phase 1: scores = (Q K^T) * 1/8, into LDS -------------------------
  Frag16 aq[2];
#pragma unroll
  for (int j = 0; j < 2; ++j) {
    const char* qb = Qh + (size_t)(qt * 16 + l16) * 128 + j * 64;
    aq[j].q[0] = *(const u32x4*)(qb + 16 * half);
    aq[j].q[1] = *(const u32x4*)(qb + 32 + 16 * half);
  }

  for (int nt = w; nt < SEQ / 16; nt += 16) {   // each wave: 8 key tiles
    v8f cs = (v8f){0.f, 0.f, 0.f, 0.f, 0.f, 0.f, 0.f, 0.f};
#pragma unroll
    for (int j = 0; j < 2; ++j) {
      Frag16 bk;
      const char* kb_ = Kh + (size_t)(nt * 16 + l16) * 128 + j * 64;
      bk.q[0] = *(const u32x4*)(kb_ + 16 * half);
      bk.q[1] = *(const u32x4*)(kb_ + 32 + 16 * half);
      cs = __builtin_amdgcn_wmma_f32_16x16x32_bf16(
          false, aq[j].v, false, bk.v, (short)0, cs, false, false);
    }
#pragma unroll
    for (int r = 0; r < 8; ++r)
      sc[(r + 8 * half) * SEQ + nt * 16 + l16] = 0.125f * cs[r];
  }
  __syncthreads();

  // ---- Phase 2: per-row exact top-k (radix select) + masked softmax ------
  // Wave w owns query row w. Keys cached in registers: LDS touched once for
  // read and once for the final attn-weight writeback.
  {
    float* row = sc + w * SEQ;
    unsigned int keys[SEQ / 32];
#pragma unroll
    for (int i = 0; i < SEQ / 32; ++i) keys[i] = mono_key(row[i * 32 + lane]);

    float thr = fminf(fmaxf(*thr_ptr, 0.f), 1.f);
    unsigned int thrkey = mono_key(thr);

    unsigned int cand = 0u;                    // radix search for kth key
    for (int bit = 31; bit >= 0; --bit) {
      unsigned int trial = cand | (1u << bit);
      int cnt = 0;
#pragma unroll
      for (int i = 0; i < SEQ / 32; ++i) cnt += (keys[i] >= trial) ? 1 : 0;
#pragma unroll
      for (int off = 16; off >= 1; off >>= 1) cnt += __shfl_xor(cnt, off, 32);
      if (cnt >= KKEEP) cand = trial;
    }
    const unsigned int keepmin = (cand > thrkey) ? cand : thrkey;

    unsigned int mk = 0u;                      // max kept key == row max
#pragma unroll
    for (int i = 0; i < SEQ / 32; ++i)
      if (keys[i] >= keepmin && keys[i] > mk) mk = keys[i];
#pragma unroll
    for (int off = 16; off >= 1; off >>= 1) {
      unsigned int o = (unsigned int)__shfl_xor((int)mk, off, 32);
      if (o > mk) mk = o;
    }
    const float rmax = key_to_f(mk);

    float rsum = 0.f;
#pragma unroll
    for (int i = 0; i < SEQ / 32; ++i)
      if (keys[i] >= keepmin) rsum += __expf(key_to_f(keys[i]) - rmax);
#pragma unroll
    for (int off = 16; off >= 1; off >>= 1) rsum += __shfl_xor(rsum, off, 32);
    const float inv = (rsum > 0.f) ? (1.f / rsum) : 0.f;

#pragma unroll
    for (int i = 0; i < SEQ / 32; ++i) {
      bool keep = keys[i] >= keepmin;
      row[i * 32 + lane] =
          keep ? __expf(key_to_f(keys[i]) - rmax) * inv : 0.f;
    }
  }
  __syncthreads();

  // ---- Phase 3: out = attn @ V via WMMA ----------------------------------
  // wave -> (d-tile, k-chunk): 4 d-tiles x 4 K splits of 512 keys each.
  const int dt = w & 3;
  const int kc = w >> 2;
  v8f co = (v8f){0.f, 0.f, 0.f, 0.f, 0.f, 0.f, 0.f, 0.f};
  const int d = dt * 16 + l16;
  for (int c32 = 0; c32 < 16; ++c32) {
    int k0 = kc * 512 + c32 * 32;
    Frag16 aa, bb;
#pragma unroll
    for (int v = 0; v < 8; ++v) {
      int kk = k0 + ((v < 4) ? (2 * v + 8 * half) : (16 + 2 * (v - 4) + 8 * half));
      aa.u[v] = pack2_bf16(sc[l16 * SEQ + kk], sc[l16 * SEQ + kk + 1]);
    }
    const char* vb = Vh + (size_t)d * SEQ * 2 + k0 * 2;
    bb.q[0] = *(const u32x4*)(vb + 16 * half);
    bb.q[1] = *(const u32x4*)(vb + 32 + 16 * half);
    co = __builtin_amdgcn_wmma_f32_16x16x32_bf16(
        false, aa.v, false, bb.v, (short)0, co, false, false);
  }
  __syncthreads();   // all LDS attn reads done; safe to reuse for partials

  float* part = sc;  // [kc][dt][r][lane] = 4*4*8*32 floats (16 KB)
#pragma unroll
  for (int r = 0; r < 8; ++r)
    part[(((kc * 4 + dt) * 8 + r) * 32) + lane] = co[r];
  __syncthreads();

  if (kc == 0) {
    int b_ = bh >> 4, h = bh & 15;
#pragma unroll
    for (int r = 0; r < 8; ++r) {
      float s = 0.f;
#pragma unroll
      for (int p = 0; p < 4; ++p)
        s += part[(((p * 4 + dt) * 8 + r) * 32) + lane];
      int m   = r + 8 * half;
      int tok = b_ * SEQ + qt * 16 + m;
      attn_out[(size_t)tok * HIDDEN + h * HEAD_DIM + dt * 16 + l16] =
          f32_to_bf16(s);
    }
  }
}

// ---------------------------------------------------------------------------
// Host-side launch orchestration (graph-capture safe: kernels only)
// ---------------------------------------------------------------------------
extern "C" void kernel_launch(void* const* d_in, const int* in_sizes, int n_in,
                              void* d_out, int out_size, void* d_ws,
                              size_t ws_size, hipStream_t stream) {
  (void)in_sizes; (void)n_in; (void)out_size; (void)ws_size;
  const float* x   = (const float*)d_in[0];
  const float* Wq  = (const float*)d_in[1];
  const float* Wk  = (const float*)d_in[2];
  const float* Wv  = (const float*)d_in[3];
  const float* Wo  = (const float*)d_in[4];
  const float* bo  = (const float*)d_in[5];
  const float* thr = (const float*)d_in[6];

  // Workspace layout (48 MB total)
  char* ws = (char*)d_ws;
  unsigned short* xb  = (unsigned short*)(ws + 0);           // 8 MB x bf16
  unsigned short* wqb = (unsigned short*)(ws + (8u  << 20)); // 2 MB each
  unsigned short* wkb = (unsigned short*)(ws + (10u << 20));
  unsigned short* wvb = (unsigned short*)(ws + (12u << 20));
  unsigned short* wob = (unsigned short*)(ws + (14u << 20));
  unsigned short* qb  = (unsigned short*)(ws + (16u << 20)); // 8 MB [B,H,S,D]
  unsigned short* kb  = (unsigned short*)(ws + (24u << 20)); // 8 MB [B,H,S,D]
  unsigned short* vtb = (unsigned short*)(ws + (32u << 20)); // 8 MB [B,H,D,S]
  unsigned short* aob = (unsigned short*)(ws + (40u << 20)); // 8 MB [B*S,1024]

  const int nX = TOKENS * HIDDEN;   // 4194304
  const int nW = HIDDEN * HIDDEN;   // 1048576
  cvt_f32_to_bf16_k<<<(nX + 255) / 256, 256, 0, stream>>>(x,  xb,  nX);
  cvt_f32_to_bf16_k<<<(nW + 255) / 256, 256, 0, stream>>>(Wq, wqb, nW);
  cvt_f32_to_bf16_k<<<(nW + 255) / 256, 256, 0, stream>>>(Wk, wkb, nW);
  cvt_f32_to_bf16_k<<<(nW + 255) / 256, 256, 0, stream>>>(Wv, wvb, nW);
  cvt_f32_to_bf16_k<<<(nW + 255) / 256, 256, 0, stream>>>(Wo, wob, nW);

  dim3 gproj(HIDDEN / 128, TOKENS / 128);   // (8, 32)
  wmma_gemm_nt_k<1><<<gproj, 256, 0, stream>>>(xb, wqb, nullptr, qb,
                                               TOKENS, HIDDEN, HIDDEN);
  wmma_gemm_nt_k<1><<<gproj, 256, 0, stream>>>(xb, wkb, nullptr, kb,
                                               TOKENS, HIDDEN, HIDDEN);
  wmma_gemm_nt_k<2><<<gproj, 256, 0, stream>>>(xb, wvb, nullptr, vtb,
                                               TOKENS, HIDDEN, HIDDEN);

  // One block per (b,h,q-tile): 2*16*128 = 4096 blocks, 128 KB dynamic LDS
  attn_sparse_k<<<BATCH * HEADS * (SEQ / 16), 512, 128 * 1024, stream>>>(
      qb, kb, vtb, thr, aob);

  wmma_gemm_nt_k<0><<<gproj, 256, 0, stream>>>(aob, wob, bo, d_out,
                                               TOKENS, HIDDEN, HIDDEN);
}
